// ExactAttention_22728966930487
// MI455X (gfx1250) — matrix operand order repaired and checked by reading, not compiled, not run on hardware
//
#include <hip/hip_runtime.h>

// ---------------------------------------------------------------------------
// Flash attention for MI455X (gfx1250, wave32), fp32 in/out.
//   B=2, N=2048, H=16, D=64.  Compute-bound -> v_wmma_f32_16x16x32_f16.
// Round-3 changes (from instruction histogram):
//   * 32 query rows per wave (BLOCK_M=128): each B-operand ds_load_b128 pair
//     feeds two WMMAs; 32 WMMAs per barrier; staging cost amortized 2x.
//   * Row-max butterfly done on f16x2-packed pairs (v_pk_max + ds_bpermute):
//     2 shuffles/row instead of 4, batched behind one s_wait_dscnt per step.
//     (m need not be exact: any m within ~15 of true max keeps P in f16 range,
//      and the same m is used consistently for exp/alpha/l -> still exact softmax.)
//   * Everything else as round-2: f16 LDS tiles (V transposed), exp2 domain,
//     deferred l reduction, double-buffered tiles, one barrier/iteration.
// ---------------------------------------------------------------------------

typedef __attribute__((ext_vector_type(16))) _Float16 v16h;
typedef __attribute__((ext_vector_type(8)))  _Float16 v8h;
typedef __attribute__((ext_vector_type(4)))  _Float16 v4h;
typedef __attribute__((ext_vector_type(2)))  _Float16 h2;
typedef __attribute__((ext_vector_type(8)))  float    v8f;
typedef __attribute__((ext_vector_type(4)))  float    v4f;

#define BB 2
#define NN 2048
#define HH 16
#define DD 64
#define BLOCK_M 128           // queries per workgroup (32 per wave, 2 M-subtiles)
#define BLOCK_N 64            // keys per iteration
#define NWAVES  4
#define LDH     72            // padded LDS row stride in halves (144B: 16B aligned, bank-safe)
#define LOG2E   1.44269504088896340736f

union V16U { v16h v; v8h h[2]; };

__global__ __launch_bounds__(128)
void ExactAttention_22728966930487_kernel(const float* __restrict__ Q,
                                          const float* __restrict__ K,
                                          const float* __restrict__ V,
                                          float* __restrict__ Out) {
    __shared__ _Float16 K16[2][BLOCK_N * LDH];    // [key][d]  f16, double buffered
    __shared__ _Float16 Vt [2][DD * LDH];         // [d][key]  f16 (transposed), double buffered
    __shared__ _Float16 Plds[NWAVES * 32 * LDH];  // per-wave P transpose buffer (32 rows)

    const int tid  = threadIdx.x;
    const int wave = tid >> 5;
    const int lane = tid & 31;
    const int lo   = lane & 15;
    const int hi   = lane >> 4;

    const int mtile = blockIdx.x;   // 0..15
    const int h     = blockIdx.y;   // 0..15
    const int b     = blockIdx.z;   // 0..1

    const int qrow0 = mtile * BLOCK_M + wave * 32;   // wave's first of 32 query rows

    // Q/K/V layout [B, N, H, D]: addr(b,n,h,d) = ((b*N+n)*H+h)*D + d ; n-stride = 1024 floats.
    const size_t rowStride = (size_t)HH * DD;
    const float* Kbase = K + (size_t)b * NN * rowStride + (size_t)h * DD;
    const float* Vbase = V + (size_t)b * NN * rowStride + (size_t)h * DD;

    // ---- Q tiles (2 x 16x64) -> WMMA A layout, f16, pre-scaled by log2(e) ----
    // A chunk c (16x32): lane = row lo; element e -> d = 32c + (e>=8?16:0) + 8*hi + (e&7).
    v16h a_q[2][2];
    #pragma unroll
    for (int u = 0; u < 2; ++u) {
        const float* qrow = Q + ((size_t)(b * NN + qrow0 + 16 * u + lo) * HH + h) * DD;
        const v4f*   q4   = (const v4f*)qrow;
        #pragma unroll
        for (int c = 0; c < 2; ++c) {
            const int f0 = (32 * c + 8 * hi) >> 2;
            v4f s0 = q4[f0],     s1 = q4[f0 + 1];
            v4f s2 = q4[f0 + 4], s3 = q4[f0 + 5];
            #pragma unroll
            for (int e = 0; e < 4; ++e) {
                a_q[u][c][e]      = (_Float16)(s0[e] * LOG2E);
                a_q[u][c][e + 4]  = (_Float16)(s1[e] * LOG2E);
                a_q[u][c][e + 8]  = (_Float16)(s2[e] * LOG2E);
                a_q[u][c][e + 12] = (_Float16)(s3[e] * LOG2E);
            }
        }
    }

    // ---- cooperative staging: K tile row-major f16, V tile transposed f16 ----
    auto stage = [&](int j, int buf) {
        #pragma unroll
        for (int i = 0; i < 8; ++i) {                  // K: 1024 float4, 8 per thread
            const int idx = tid + 128 * i;
            const int row = idx >> 4;
            const int c4  = (idx & 15) << 2;
            const float* src = Kbase + (size_t)(j + row) * rowStride + c4;
            v4f kv = *(const v4f*)src;
            v4h kh;
            #pragma unroll
            for (int q = 0; q < 4; ++q) kh[q] = (_Float16)kv[q];
            *(v4h*)&K16[buf][row * LDH + c4] = kh;
            if (j + 3 * BLOCK_N <= NN)                 // global_prefetch_b8 for tile after next
                __builtin_prefetch(src + 2 * BLOCK_N * rowStride, 0, 1);
        }
        #pragma unroll
        for (int i = 0; i < 4; ++i) {                  // V: transpose 2 keys x 4 d per unit
            const int u    = tid + 128 * i;
            const int row2 = (u >> 4) << 1;
            const int c4   = (u & 15) << 2;
            const float* s0 = Vbase + (size_t)(j + row2) * rowStride + c4;
            v4f va = *(const v4f*)s0;
            v4f vb = *(const v4f*)(s0 + rowStride);
            #pragma unroll
            for (int q = 0; q < 4; ++q) {
                h2 pk; pk[0] = (_Float16)va[q]; pk[1] = (_Float16)vb[q];
                *(h2*)&Vt[buf][(c4 + q) * LDH + row2] = pk;
            }
        }
    };

    // ---- flash accumulators (2 M-subtiles) ----
    v8f   acc[2][4];
    float m_row[2][8], l_row[2][8];
    #pragma unroll
    for (int u = 0; u < 2; ++u) {
        #pragma unroll
        for (int t = 0; t < 4; ++t) acc[u][t] = (v8f){0.f,0.f,0.f,0.f,0.f,0.f,0.f,0.f};
        #pragma unroll
        for (int r = 0; r < 8; ++r) { m_row[u][r] = -1.0e30f; l_row[u][r] = 0.f; }
    }

    _Float16* pb = &Plds[wave * 32 * LDH];

    stage(0, 0);
    int buf = 0;

    #pragma unroll 1
    for (int j = 0; j < NN; j += BLOCK_N) {
        __syncthreads();                                      // staged tile visible; prev compute done
        if (j + BLOCK_N < NN) stage(j + BLOCK_N, buf ^ 1);    // overlap with compute below

        // ---- S = (Q*log2e) * K^T : 2 subtiles x 4 tiles; each bk feeds 2 WMMAs ----
        v8f s[2][4];
        #pragma unroll
        for (int u = 0; u < 2; ++u)
            #pragma unroll
            for (int t = 0; t < 4; ++t) s[u][t] = (v8f){0.f,0.f,0.f,0.f,0.f,0.f,0.f,0.f};
        #pragma unroll
        for (int t = 0; t < 4; ++t) {
            #pragma unroll
            for (int c = 0; c < 2; ++c) {
                const _Float16* kp = &K16[buf][(16 * t + lo) * LDH + 32 * c + 8 * hi];
                V16U bk; bk.h[0] = *(const v8h*)kp; bk.h[1] = *(const v8h*)(kp + 16);
                #pragma unroll
                for (int u = 0; u < 2; ++u)
                    s[u][t] = __builtin_amdgcn_wmma_f32_16x16x32_f16(
                        false, a_q[u][c], false, bk.v, (short)0, s[u][t], false, false);
            }
        }

        // ---- online softmax, base-2 domain ----
        // Local per-row max over the 4 tiles (f32), then cross-lane butterfly on
        // f16x2-packed row pairs: 8 regs x 4 steps, v_pk_max, waits batched.
        float lm[2][8];
        #pragma unroll
        for (int u = 0; u < 2; ++u)
            #pragma unroll
            for (int r = 0; r < 8; ++r)
                lm[u][r] = fmaxf(fmaxf(s[u][0][r], s[u][1][r]),
                                 fmaxf(s[u][2][r], s[u][3][r]));
        h2 pkm[8];
        #pragma unroll
        for (int k = 0; k < 8; ++k) {
            const int u = k >> 2, rr = (k & 3) << 1;
            pkm[k][0] = (_Float16)lm[u][rr];
            pkm[k][1] = (_Float16)lm[u][rr + 1];
        }
        #pragma unroll
        for (int mask = 1; mask <= 8; mask <<= 1) {
            #pragma unroll
            for (int k = 0; k < 8; ++k) {
                const int sv = __shfl_xor(__builtin_bit_cast(int, pkm[k]), mask, 32);
                pkm[k] = __builtin_elementwise_max(pkm[k], __builtin_bit_cast(h2, sv));
            }
        }
        float alpha[2][8];
        #pragma unroll
        for (int u = 0; u < 2; ++u) {
            #pragma unroll
            for (int r = 0; r < 8; ++r) {
                const float mx   = (float)pkm[(u << 2) + (r >> 1)][r & 1];
                const float mnew = fmaxf(m_row[u][r], mx);
                const float a    = __builtin_amdgcn_exp2f(m_row[u][r] - mnew);
                alpha[u][r] = a;  m_row[u][r] = mnew;
                float ps = 0.f;
                #pragma unroll
                for (int t = 0; t < 4; ++t) {
                    const float p = __builtin_amdgcn_exp2f(s[u][t][r] - mnew);
                    s[u][t][r] = p;  ps += p;
                }
                l_row[u][r] = l_row[u][r] * a + ps;   // per-lane partial; reduced after loop
            }
        }
        #pragma unroll
        for (int u = 0; u < 2; ++u)
            #pragma unroll
            for (int t = 0; t < 4; ++t)
                #pragma unroll
                for (int r = 0; r < 8; ++r) acc[u][t][r] *= alpha[u][r];

        // ---- transpose P (C layout -> A layout) through per-wave LDS, f16 ----
        #pragma unroll
        for (int u = 0; u < 2; ++u)
            #pragma unroll
            for (int t = 0; t < 4; ++t)
                #pragma unroll
                for (int r = 0; r < 8; ++r)
                    pb[(16 * u + r + 8 * hi) * LDH + 16 * t + lo] = (_Float16)s[u][t][r];

        v16h pa[2][2];  // A: chunk c element e -> key = 32c + (e>=8?16:0) + 8*hi + (e&7)
        #pragma unroll
        for (int u = 0; u < 2; ++u)
            #pragma unroll
            for (int c = 0; c < 2; ++c) {
                const _Float16* pp = &pb[(16 * u + lo) * LDH + 32 * c + 8 * hi];
                V16U w; w.h[0] = *(const v8h*)pp; w.h[1] = *(const v8h*)(pp + 16);
                pa[u][c] = w.v;
            }

        // ---- O += P * V : each bv feeds 2 WMMAs ----
        #pragma unroll
        for (int t = 0; t < 4; ++t) {
            #pragma unroll
            for (int c = 0; c < 2; ++c) {
                const _Float16* vp = &Vt[buf][(16 * t + lo) * LDH + 32 * c + 8 * hi];
                V16U bv; bv.h[0] = *(const v8h*)vp; bv.h[1] = *(const v8h*)(vp + 16);
                #pragma unroll
                for (int u = 0; u < 2; ++u)
                    acc[u][t] = __builtin_amdgcn_wmma_f32_16x16x32_f16(
                        false, pa[u][c], false, bv.v, (short)0, acc[u][t], false, false);
            }
        }
        buf ^= 1;
    }

    // ---- epilogue: finish deferred l reduction (f32), normalize, store [B,H,N,D] ----
    float* obase = Out + ((size_t)(b * HH + h) * NN) * DD;
    #pragma unroll
    for (int u = 0; u < 2; ++u) {
        float inv[8];
        #pragma unroll
        for (int r = 0; r < 8; ++r) {
            float l = l_row[u][r];
            l += __shfl_xor(l, 1, 32);
            l += __shfl_xor(l, 2, 32);
            l += __shfl_xor(l, 4, 32);
            l += __shfl_xor(l, 8, 32);
            inv[r] = 1.0f / l;
        }
        #pragma unroll
        for (int t = 0; t < 4; ++t)
            #pragma unroll
            for (int r = 0; r < 8; ++r) {
                const int n = qrow0 + 16 * u + r + 8 * hi;
                obase[(size_t)n * DD + 16 * t + lo] = acc[u][t][r] * inv[r];
            }
    }
}

extern "C" void kernel_launch(void* const* d_in, const int* in_sizes, int n_in,
                              void* d_out, int out_size, void* d_ws, size_t ws_size,
                              hipStream_t stream) {
    (void)in_sizes; (void)n_in; (void)d_ws; (void)ws_size; (void)out_size;
    const float* Q = (const float*)d_in[0];
    const float* K = (const float*)d_in[1];
    const float* V = (const float*)d_in[2];
    float*       O = (float*)d_out;
    dim3 grid(NN / BLOCK_M, HH, BB);   // (16, 16, 2)
    ExactAttention_22728966930487_kernel<<<grid, dim3(128), 0, stream>>>(Q, K, V, O);
}
